// GraphTransformerLayer_31971736551808
// MI455X (gfx1250) — matrix operand order repaired and checked by reading, not compiled, not run on hardware
//
#include <hip/hip_runtime.h>
#include <hip/hip_bf16.h>

// ---------------------------------------------------------------------------
// GraphTransformerLayer for MI455X (gfx1250, wave32, WMMA f16 path)
// ---------------------------------------------------------------------------

typedef __attribute__((ext_vector_type(16))) _Float16 v16h;
typedef __attribute__((ext_vector_type(8)))  _Float16 v8h;
typedef __attribute__((ext_vector_type(4)))  _Float16 v4h;
typedef __attribute__((ext_vector_type(8)))  float    v8f;
typedef __attribute__((ext_vector_type(4)))  float    v4f;

#define HID   1024
#define FFN   2048
#define SEQ   2048
#define BSZ   2
#define NHEAD 16
#define HDIM  64
#define MROWS (BSZ * SEQ)   // 4096 flattened (b, n) rows

// Build a 16-wide f16 fragment from two contiguous 8-element (16B) chunks.
__device__ __forceinline__ v16h ldfrag(const _Float16* p0, const _Float16* p1) {
  v8h lo = *(const v8h*)p0;
  v8h hi = *(const v8h*)p1;
  return __builtin_shufflevector(lo, hi, 0, 1, 2, 3, 4, 5, 6, 7,
                                 8, 9, 10, 11, 12, 13, 14, 15);
}

__device__ __forceinline__ v8f wmma_f16(v16h a, v16h b, v8f c) {
  return __builtin_amdgcn_wmma_f32_16x16x32_f16(false, a, false, b,
                                                (short)0, c, false, false);
}

// ---------------------------------------------------------------------------
// f32 -> f16 weight convert + transpose:  Wt[n][k] = (f16) W[k][n]
// ---------------------------------------------------------------------------
__global__ void wt_cvt_kernel(const float* __restrict__ W,
                              _Float16* __restrict__ Wt, int K, int N) {
  long long idx = (long long)blockIdx.x * 256 + threadIdx.x;
  if (idx >= (long long)K * N) return;
  int n = (int)(idx / K);
  int k = (int)(idx % K);
  Wt[idx] = (_Float16)W[(long long)k * N + n];
}

// ---------------------------------------------------------------------------
// rel_bias transpose+downconvert: [B][N][N][H] f32 -> [B][H][N][N] f16.
// Makes the dominant (536 MB) stream fully coalesced in the attention kernel
// instead of a 64B-strided 4B gather (up to 16x HBM amplification).
// ---------------------------------------------------------------------------
__global__ void __launch_bounds__(256)
bias_tr_kernel(const float* __restrict__ rb, _Float16* __restrict__ rbt) {
  __shared__ _Float16 tile[NHEAD][64 + 4];
  const int kv0 = blockIdx.x * 64;
  const int q   = blockIdx.y;
  const int b   = blockIdx.z;
  const int tid = threadIdx.x;

  // load 64 kv x 16 h; consecutive tid -> h fastest (contiguous 64B per 16 thr)
  const int h_in  = tid & 15;
  const int kv_in = tid >> 4;  // 0..15
  const float* src = rb + ((long long)(b * SEQ + q) * SEQ + kv0) * NHEAD;
#pragma unroll
  for (int r = 0; r < 4; ++r) {
    const int kv = kv_in + r * 16;
    tile[h_in][kv] = (_Float16)src[(long long)kv * NHEAD + h_in];
  }
  __syncthreads();

  // store; consecutive tid -> kv fastest (contiguous 128B per 64 thr)
  const int kv_out = tid & 63;
  const int h_out  = tid >> 6;  // 0..3
#pragma unroll
  for (int r = 0; r < 4; ++r) {
    const int h = h_out + r * 4;
    rbt[((long long)(b * NHEAD + h) * SEQ + q) * SEQ + kv0 + kv_out] =
        tile[h][kv_out];
  }
}

// ---------------------------------------------------------------------------
// LayerNorm over rows of 1024, f32 in -> f16 out (WMMA A operand precision)
// ---------------------------------------------------------------------------
__global__ void __launch_bounds__(256)
ln_kernel(const float* __restrict__ x, const float* __restrict__ g,
          const float* __restrict__ b, _Float16* __restrict__ out) {
  __shared__ float red[256];
  const int row = blockIdx.x;
  const int tid = threadIdx.x;
  const float* xr = x + (long long)row * HID;

  v4f xv = *(const v4f*)&xr[tid * 4];
  float s = xv.x + xv.y + xv.z + xv.w;
  red[tid] = s;
  __syncthreads();
  for (int st = 128; st > 0; st >>= 1) {
    if (tid < st) red[tid] += red[tid + st];
    __syncthreads();
  }
  const float mu = red[0] * (1.0f / (float)HID);
  __syncthreads();

  float d0 = xv.x - mu, d1 = xv.y - mu, d2 = xv.z - mu, d3 = xv.w - mu;
  red[tid] = d0 * d0 + d1 * d1 + d2 * d2 + d3 * d3;
  __syncthreads();
  for (int st = 128; st > 0; st >>= 1) {
    if (tid < st) red[tid] += red[tid + st];
    __syncthreads();
  }
  const float var = red[0] * (1.0f / (float)HID);
  const float inv = rsqrtf(var + 1e-5f);

  v4f gv = *(const v4f*)&g[tid * 4];
  v4f bv = *(const v4f*)&b[tid * 4];
  v4h ov;
  ov.x = (_Float16)(d0 * inv * gv.x + bv.x);
  ov.y = (_Float16)(d1 * inv * gv.y + bv.y);
  ov.z = (_Float16)(d2 * inv * gv.z + bv.z);
  ov.w = (_Float16)(d3 * inv * gv.w + bv.w);
  *(v4h*)&out[(long long)row * HID + tid * 4] = ov;
}

// ---------------------------------------------------------------------------
// WMMA GEMM:  C[M, Nout] = A_f16[M, K] @ Wt_f16[Nout, K]^T + bias
// 64x64 tile per 256-thread block (8 waves, 2 C-tiles each), K staged in LDS.
// Epilogue modes: RELU, residual add (f32), f16/f32 out, V-transpose layout.
// ---------------------------------------------------------------------------
#define LPAD 40  // 32 f16 + 8 pad -> 80B row stride (bank-conflict free 16B ds)

template <bool RELU, bool RESID, bool OUT16, bool VTRANS>
__global__ void __launch_bounds__(256)
gemm_kernel(const _Float16* __restrict__ A, const _Float16* __restrict__ Bt,
            const float* __restrict__ bias, const float* __restrict__ resid,
            float* __restrict__ outF, _Float16* __restrict__ outH,
            int K, int Nout) {
  __shared__ _Float16 ldsA[64 * LPAD];
  __shared__ _Float16 ldsB[64 * LPAD];

  const int tid  = threadIdx.x;
  const int lane = tid & 31;
  const int wave = tid >> 5;
  const int l15  = lane & 15;
  const int hi   = lane >> 4;
  const int msub = wave & 3;   // which 16-row strip of the 64-row tile
  const int npr  = wave >> 2;  // which pair of 16-col strips
  const int m0 = blockIdx.y * 64;
  const int n0 = blockIdx.x * 64;

  const int lrow = tid >> 2;        // 0..63 cooperative-load row
  const int lcol = (tid & 3) * 8;   // 0,8,16,24

  v8f acc0 = {};
  v8f acc1 = {};

  for (int k0 = 0; k0 < K; k0 += 32) {
    *(v8h*)&ldsA[lrow * LPAD + lcol] =
        *(const v8h*)&A[(long long)(m0 + lrow) * K + k0 + lcol];
    *(v8h*)&ldsB[lrow * LPAD + lcol] =
        *(const v8h*)&Bt[(long long)(n0 + lrow) * K + k0 + lcol];
    __syncthreads();

    const _Float16* pa = &ldsA[(msub * 16 + l15) * LPAD + hi * 16];
    v16h a = ldfrag(pa, pa + 8);
    const _Float16* pb0 = &ldsB[((npr * 2 + 0) * 16 + l15) * LPAD + hi * 16];
    const _Float16* pb1 = &ldsB[((npr * 2 + 1) * 16 + l15) * LPAD + hi * 16];
    v16h b0 = ldfrag(pb0, pb0 + 8);
    v16h b1 = ldfrag(pb1, pb1 + 8);

    acc0 = wmma_f16(a, b0, acc0);
    acc1 = wmma_f16(a, b1, acc1);
    __syncthreads();
  }

#pragma unroll
  for (int t = 0; t < 2; ++t) {
    v8f acc = t ? acc1 : acc0;
    const int n = n0 + (npr * 2 + t) * 16 + l15;
    const float bv = bias[n];
#pragma unroll
    for (int v = 0; v < 8; ++v) {
      const int m = m0 + msub * 16 + hi * 8 + v;
      float val = acc[v] + bv;
      if (RELU) val = val > 0.0f ? val : 0.0f;
      if (RESID) val += resid[(long long)m * Nout + n];
      if (OUT16) {
        if (VTRANS) {
          // write V as Vt[b][h][d][seq] so PV B-fragments are contiguous
          const int bb = m >> 11, seq = m & 2047;
          const int hh = n >> 6, dd = n & 63;
          outH[(((long long)(bb * NHEAD + hh) * HDIM + dd) << 11) + seq] =
              (_Float16)val;
        } else {
          outH[(long long)m * Nout + n] = (_Float16)val;
        }
      } else {
        outF[(long long)m * Nout + n] = val;
      }
    }
  }
}

// ---------------------------------------------------------------------------
// Flash-style attention: one wave per 16 query rows of one (b, h).
// Streams K/V in 32-kv steps; transposed f16 rel_bias fused into score tiles;
// online softmax; prefetch of next-tile K/V/bias streams.
// Q, K row-major f16 [B*N][1024]; V transposed f16 [B][H][64][N];
// bias transposed f16 [B][H][N][N].
// ---------------------------------------------------------------------------
__global__ void __launch_bounds__(256)
attn_kernel(const _Float16* __restrict__ Q, const _Float16* __restrict__ Km,
            const _Float16* __restrict__ Vt, const _Float16* __restrict__ rbt,
            _Float16* __restrict__ O) {
  __shared__ _Float16 pbuf[8][16 * LPAD];  // per-wave P transpose buffer

  const int tid  = threadIdx.x;
  const int lane = tid & 31;
  const int wave = tid >> 5;
  const int l15  = lane & 15;
  const int hi   = lane >> 4;
  const int h = blockIdx.y;
  const int b = blockIdx.z;
  const int qbase = (blockIdx.x * 8 + wave) * 16;

  // Q fragments: 16 rows x 64 k, two A-fragments (k 0..31, 32..63)
  const _Float16* qp =
      Q + (long long)(b * SEQ + qbase + l15) * HID + h * HDIM;
  v16h a0 = ldfrag(qp + hi * 16, qp + hi * 16 + 8);
  v16h a1 = ldfrag(qp + 32 + hi * 16, qp + 40 + hi * 16);

  v8f o[4] = {v8f{}, v8f{}, v8f{}, v8f{}};
  float rm[8], rs[8];
#pragma unroll
  for (int v = 0; v < 8; ++v) { rm[v] = -1e30f; rs[v] = 0.0f; }

  const _Float16* vbase = Vt + (long long)(b * NHEAD + h) * HDIM * SEQ;
  // transposed bias rows for this wave's 8 q rows (per half-wave group)
  const _Float16* bbase = rbt + ((long long)(b * NHEAD + h) * SEQ +
                                 (qbase + hi * 8)) * SEQ;

  for (int j = 0; j < SEQ / 32; ++j) {
    const int kv0 = j * 32;

    // --- scores S = (Q K^T) * 0.125 + rel_bias : two 16x16 tiles ---
    const _Float16* kp0 =
        Km + (long long)(b * SEQ + kv0 + l15) * HID + h * HDIM;
    const _Float16* kp1 = kp0 + 16 * HID;
    v16h bk0 = ldfrag(kp0 + hi * 16, kp0 + hi * 16 + 8);        // k 0..31
    v16h bk1 = ldfrag(kp0 + 32 + hi * 16, kp0 + 40 + hi * 16);  // k 32..63
    v16h bk2 = ldfrag(kp1 + hi * 16, kp1 + hi * 16 + 8);
    v16h bk3 = ldfrag(kp1 + 32 + hi * 16, kp1 + 40 + hi * 16);

    // prefetch next kv tile's K rows (global_prefetch_b8)
    if (j + 1 < SEQ / 32) {
      __builtin_prefetch(kp0 + 32 * HID, 0, 1);
      __builtin_prefetch(kp1 + 32 * HID, 0, 1);
    }

    v8f s0 = {}, s1 = {};
    s0 = wmma_f16(a0, bk0, s0);
    s0 = wmma_f16(a1, bk1, s0);
    s1 = wmma_f16(a0, bk2, s1);
    s1 = wmma_f16(a1, bk3, s1);

    float mx[8];
#pragma unroll
    for (int v = 0; v < 8; ++v) {
      const _Float16* br = bbase + (long long)v * SEQ + kv0;
      const float b0v = (float)br[l15];        // coalesced: lane -> kv
      const float b1v = (float)br[16 + l15];
      __builtin_prefetch(br + 128, 0, 1);      // 4 tiles ahead in the kv sweep
      s0[v] = s0[v] * 0.125f + b0v;
      s1[v] = s1[v] * 0.125f + b1v;
      mx[v] = fmaxf(s0[v], s1[v]);
    }
    // row max across the 16 column-lanes of each half-wave group
#pragma unroll
    for (int off = 1; off < 16; off <<= 1)
#pragma unroll
      for (int v = 0; v < 8; ++v)
        mx[v] = fmaxf(mx[v], __shfl_xor(mx[v], off, 32));

    float ps[8], scl[8];
#pragma unroll
    for (int v = 0; v < 8; ++v) {
      const float nm = fmaxf(rm[v], mx[v]);
      scl[v] = __expf(rm[v] - nm);
      rm[v] = nm;
      const float p0 = __expf(s0[v] - nm);
      const float p1 = __expf(s1[v] - nm);
      s0[v] = p0;
      s1[v] = p1;
      ps[v] = p0 + p1;
    }
#pragma unroll
    for (int off = 1; off < 16; off <<= 1)
#pragma unroll
      for (int v = 0; v < 8; ++v) ps[v] += __shfl_xor(ps[v], off, 32);
#pragma unroll
    for (int v = 0; v < 8; ++v) {
      rs[v] = rs[v] * scl[v] + ps[v];
      o[0][v] *= scl[v];
      o[1][v] *= scl[v];
      o[2][v] *= scl[v];
      o[3][v] *= scl[v];
    }

    // --- transpose P (C layout -> A layout) through per-wave LDS ---
    _Float16* pb = pbuf[wave];
#pragma unroll
    for (int v = 0; v < 8; ++v) {
      pb[(hi * 8 + v) * LPAD + l15] = (_Float16)s0[v];
      pb[(hi * 8 + v) * LPAD + 16 + l15] = (_Float16)s1[v];
    }
    __syncthreads();
    const _Float16* pr = &pb[l15 * LPAD + hi * 16];
    v16h pa = ldfrag(pr, pr + 8);
    __syncthreads();

    // --- O += P @ V : four 16x16 head-dim tiles, contiguous Vt reads ---
#pragma unroll
    for (int g = 0; g < 4; ++g) {
      const _Float16* vr = vbase + (long long)(g * 16 + l15) * SEQ + kv0;
      v16h bv = ldfrag(vr + hi * 16, vr + hi * 16 + 8);
      __builtin_prefetch(vr + 128, 0, 1);  // 4 tiles ahead
      o[g] = wmma_f16(pa, bv, o[g]);
    }
  }

  // epilogue: normalize and write attn_out[b*N+q][h*64+d] as f16
#pragma unroll
  for (int g = 0; g < 4; ++g) {
#pragma unroll
    for (int v = 0; v < 8; ++v) {
      const int q = qbase + hi * 8 + v;
      const int d = g * 16 + l15;
      const float val = o[g][v] / rs[v];
      O[(long long)(b * SEQ + q) * HID + h * HDIM + d] = (_Float16)val;
    }
  }
}

// ---------------------------------------------------------------------------
// Launch
// ---------------------------------------------------------------------------
extern "C" void kernel_launch(void* const* d_in, const int* in_sizes, int n_in,
                              void* d_out, int out_size, void* d_ws,
                              size_t ws_size, hipStream_t stream) {
  const float* x    = (const float*)d_in[0];
  const float* relb = (const float*)d_in[1];
  const float* Wq   = (const float*)d_in[2];
  const float* bq   = (const float*)d_in[3];
  const float* Wk   = (const float*)d_in[4];
  const float* bk   = (const float*)d_in[5];
  const float* Wv   = (const float*)d_in[6];
  const float* bv   = (const float*)d_in[7];
  const float* Wo   = (const float*)d_in[8];
  const float* bo   = (const float*)d_in[9];
  const float* ln1g = (const float*)d_in[10];
  const float* ln1b = (const float*)d_in[11];
  const float* ln2g = (const float*)d_in[12];
  const float* ln2b = (const float*)d_in[13];
  const float* W1   = (const float*)d_in[14];
  const float* b1   = (const float*)d_in[15];
  const float* W2   = (const float*)d_in[16];
  const float* b2   = (const float*)d_in[17];
  float* out = (float*)d_out;

  char* ws = (char*)d_ws;
  size_t off = 0;
  auto alloc = [&](size_t bytes) -> void* {
    void* p = ws + off;
    off += (bytes + 255) & ~(size_t)255;
    return p;
  };
  _Float16* rbt = (_Float16*)alloc((size_t)BSZ * NHEAD * SEQ * SEQ * 2);
  _Float16* h1  = (_Float16*)alloc((size_t)MROWS * HID * 2);
  _Float16* Wqt = (_Float16*)alloc((size_t)HID * HID * 2);
  _Float16* Wkt = (_Float16*)alloc((size_t)HID * HID * 2);
  _Float16* Wvt = (_Float16*)alloc((size_t)HID * HID * 2);
  _Float16* Wot = (_Float16*)alloc((size_t)HID * HID * 2);
  _Float16* W1t = (_Float16*)alloc((size_t)HID * FFN * 2);
  _Float16* W2t = (_Float16*)alloc((size_t)FFN * HID * 2);
  _Float16* Qf  = (_Float16*)alloc((size_t)MROWS * HID * 2);
  _Float16* Kf  = (_Float16*)alloc((size_t)MROWS * HID * 2);
  _Float16* Vtr = (_Float16*)alloc((size_t)MROWS * HID * 2);
  _Float16* aout = (_Float16*)alloc((size_t)MROWS * HID * 2);
  float*    x2   = (float*)alloc((size_t)MROWS * HID * 4);
  _Float16* h2   = (_Float16*)alloc((size_t)MROWS * HID * 2);
  _Float16* f1   = (_Float16*)alloc((size_t)MROWS * FFN * 2);
  (void)ws_size; (void)n_in; (void)in_sizes; (void)out_size;

  // Weight convert + transpose (once per launch; ~16 MB of f16 weights)
  wt_cvt_kernel<<<(HID * HID + 255) / 256, 256, 0, stream>>>(Wq, Wqt, HID, HID);
  wt_cvt_kernel<<<(HID * HID + 255) / 256, 256, 0, stream>>>(Wk, Wkt, HID, HID);
  wt_cvt_kernel<<<(HID * HID + 255) / 256, 256, 0, stream>>>(Wv, Wvt, HID, HID);
  wt_cvt_kernel<<<(HID * HID + 255) / 256, 256, 0, stream>>>(Wo, Wot, HID, HID);
  wt_cvt_kernel<<<(HID * FFN + 255) / 256, 256, 0, stream>>>(W1, W1t, HID, FFN);
  wt_cvt_kernel<<<(FFN * HID + 255) / 256, 256, 0, stream>>>(W2, W2t, FFN, HID);

  // rel_bias -> [B][H][N][N] f16 (dominant-traffic stream made coalesced)
  bias_tr_kernel<<<dim3(SEQ / 64, SEQ, BSZ), 256, 0, stream>>>(relb, rbt);

  // LN1
  ln_kernel<<<MROWS, 256, 0, stream>>>(x, ln1g, ln1b, h1);

  // Q/K/V projections (f16 out; V written transposed for PV fragments)
  dim3 gqkv(HID / 64, MROWS / 64);
  gemm_kernel<false, false, true, false><<<gqkv, 256, 0, stream>>>(
      h1, Wqt, bq, nullptr, nullptr, Qf, HID, HID);
  gemm_kernel<false, false, true, false><<<gqkv, 256, 0, stream>>>(
      h1, Wkt, bk, nullptr, nullptr, Kf, HID, HID);
  gemm_kernel<false, false, true, true><<<gqkv, 256, 0, stream>>>(
      h1, Wvt, bv, nullptr, nullptr, Vtr, HID, HID);

  // Flash attention with fused (transposed) rel_bias
  attn_kernel<<<dim3(SEQ / 128, NHEAD, BSZ), 256, 0, stream>>>(
      Qf, Kf, Vtr, rbt, aout);

  // Output projection + residual -> x2 (f32)
  gemm_kernel<false, true, false, false><<<gqkv, 256, 0, stream>>>(
      aout, Wot, bo, x, x2, nullptr, HID, HID);

  // LN2
  ln_kernel<<<MROWS, 256, 0, stream>>>(x2, ln2g, ln2b, h2);

  // FFN
  dim3 gffn1(FFN / 64, MROWS / 64);
  gemm_kernel<true, false, true, false><<<gffn1, 256, 0, stream>>>(
      h2, W1t, b1, nullptr, nullptr, f1, HID, FFN);
  gemm_kernel<false, true, false, false><<<gqkv, 256, 0, stream>>>(
      f1, W2t, b2, x2, out, nullptr, FFN, HID);
}
